// MultiHeadAttention_v2_41102837023258
// MI455X (gfx1250) — matrix operand order repaired and checked by reading, not compiled, not run on hardware
//
#include <hip/hip_runtime.h>
#include <hip/hip_bf16.h>

#define D_MODEL 1024
#define SEQ     2048
#define BATCH   2
#define NHEAD   16
#define HDIM    64
#define MROWS   (BATCH * SEQ)   // 4096

typedef __bf16 bf16;
typedef __attribute__((ext_vector_type(16))) __bf16 v16bf;
typedef __attribute__((ext_vector_type(8)))  __bf16 v8bf;
typedef __attribute__((ext_vector_type(8)))  float  v8f;
typedef __attribute__((ext_vector_type(4)))  unsigned int u32x4;
typedef __attribute__((ext_vector_type(4)))  int          i32x4;
typedef __attribute__((ext_vector_type(8)))  int          i32x8;

// fp32 -> bf16 with round-to-nearest-even
static __device__ __forceinline__ bf16 f2bf(float x) {
  unsigned u = __builtin_bit_cast(unsigned, x);
  unsigned r = u + 0x7FFFu + ((u >> 16) & 1u);
  unsigned short h = (unsigned short)(r >> 16);
  return __builtin_bit_cast(bf16, h);
}

// A-fragment (16x32 bf16, MxK) per ISA 7.12.2: lane holds row m=lane&15.
// lanes 0-15:  elems 0..7 <-> K = k0+0..7,   elems 8..15 <-> K = k0+16..23
// lanes 16-31: elems 0..7 <-> K = k0+8..15,  elems 8..15 <-> K = k0+24..31
// caller passes base = &row[k0 + hf*8]; we read [0..7] and [16..23].
static __device__ __forceinline__ v16bf load_frag_a(const bf16* base) {
  v8bf lo = *(const v8bf*)(base);
  v8bf hi = *(const v8bf*)(base + 16);
  v16bf r;
#pragma unroll
  for (int i = 0; i < 8; ++i) { r[i] = lo[i]; r[i + 8] = hi[i]; }
  return r;
}

// ---------------------------------------------------------------------------
// Elementwise prep kernels
// ---------------------------------------------------------------------------
__global__ void cvt_bf16_kernel(const float* __restrict__ src,
                                bf16* __restrict__ dst, int n) {
  int i = blockIdx.x * blockDim.x + threadIdx.x;
  if (i < n) dst[i] = f2bf(src[i]);
}

// W is [K][N] row-major fp32; produce Wt [N][K] bf16 (N-major) so that
// WMMA B-fragments (fixed column, K contiguous) are contiguous loads.
__global__ void packw_kernel(const float* __restrict__ W,
                             bf16* __restrict__ Wt, int K, int N) {
  int i = blockIdx.x * blockDim.x + threadIdx.x;   // i = n*K + k
  if (i >= K * N) return;
  int k = i % K;
  int n = i / K;
  Wt[i] = f2bf(W[(size_t)k * N + n]);
}

// V in gemm-output layout [b*SEQ + s][h*64 + d] -> Vt[b][h][d][s]
__global__ void transpose_v_kernel(const bf16* __restrict__ V,
                                   bf16* __restrict__ Vt) {
  int i = blockIdx.x * blockDim.x + threadIdx.x;
  if (i >= MROWS * D_MODEL) return;
  int col = i & (D_MODEL - 1);
  int s   = (i >> 10) & (SEQ - 1);
  int b   = i >> 21;                 // 1024*2048 = 2^21
  int h = col >> 6, d = col & 63;
  Vt[((size_t)(b * NHEAD + h) * HDIM + d) * SEQ + s] = V[i];
}

// ---------------------------------------------------------------------------
// Tiled bf16 WMMA GEMM: C[M][N] = A[M][K] * Bt[N][K]^T (+ bias)
// Block = 128 threads (4 waves); block tile 128x64; wave tile 32x64.
// The 64-col x K weight panel (128 KB bf16) is staged once per block into
// LDS via the Tensor Data Mover and re-used by all 4 waves for the whole
// K loop (B-fragment loads become ds_load_b128).
// ---------------------------------------------------------------------------
__global__ __launch_bounds__(128)
void gemm_bf16_wmma(const bf16* __restrict__ A, const bf16* __restrict__ Bt,
                    bf16* __restrict__ Cbf, float* __restrict__ Cf,
                    const float* __restrict__ bias, int M, int N, int K) {
  extern __shared__ char smem_raw[];
  bf16* Bs = (bf16*)smem_raw;                 // [64][K] bf16 panel

  const int lane = threadIdx.x & 31;
  const int wave = threadIdx.x >> 5;
  const int hf   = lane >> 4;
  const int ln   = lane & 15;
  const int nbase = blockIdx.x * 64;
  const int mbase = blockIdx.y * 128 + wave * 32;

  // ---- stage B panel: Bt[nbase .. nbase+63][0..K) -> LDS ----------------
#if __has_builtin(__builtin_amdgcn_tensor_load_to_lds)
  if (wave == 0) {
    unsigned long long ga =
        (unsigned long long)(uintptr_t)(Bt + (size_t)nbase * K);
    unsigned int lds_base = (unsigned int)(uintptr_t)(void*)smem_raw;

    // D# group 0: count=1 | lds_addr | global_addr[56:0] | type=2
    u32x4 g0;
    g0[0] = 1u;                                       // count = 1 valid desc
    g0[1] = lds_base;                                 // lds_addr
    g0[2] = (unsigned int)ga;                         // global_addr[31:0]
    g0[3] = ((unsigned int)(ga >> 32) & 0x01FFFFFFu)  // global_addr[56:32]
            | 0x80000000u;                            // type = 2 ("image")
    // D# group 1: data_size=2B, tensor 1024xN, tile Kx64, stride K
    i32x8 g1;
    g1[0] = 0x00010000;                               // data_size = 1 (2B)
    g1[1] = (int)(((unsigned)K & 0xFFFFu) << 16);     // tensor_dim0 lo16
    g1[2] = (int)((((unsigned)N & 0xFFFFu) << 16) |   // tensor_dim1 lo16
                  (((unsigned)K >> 16) & 0xFFFFu));   // tensor_dim0 hi16
    g1[3] = (int)((((unsigned)K & 0xFFFFu) << 16) |   // tile_dim0 = K
                  (((unsigned)N >> 16) & 0xFFFFu));   // tensor_dim1 hi16
    g1[4] = 64;                                       // tile_dim1 = 64 rows
    g1[5] = K;                                        // tensor_dim0_stride lo32
    g1[6] = 0;
    g1[7] = 0;
    i32x4 g2 = {0, 0, 0, 0};
    i32x4 g3 = {0, 0, 0, 0};
#if __clang_major__ >= 23
    i32x8 g4 = {0, 0, 0, 0, 0, 0, 0, 0};
    __builtin_amdgcn_tensor_load_to_lds(g0, g1, g2, g3, g4, 0);
#else
    __builtin_amdgcn_tensor_load_to_lds(g0, g1, g2, g3, 0);
#endif
    __builtin_amdgcn_s_wait_tensorcnt(0);
  }
  __syncthreads();
#else
  {
    const uint4* src = (const uint4*)(Bt + (size_t)nbase * K);
    uint4* dst = (uint4*)Bs;
    const int n16 = 64 * K / 8;                       // uint4 = 8 bf16
    for (int i = threadIdx.x; i < n16; i += blockDim.x) dst[i] = src[i];
    __syncthreads();
  }
#endif

  v8f acc[2][4] = {};

  for (int k0 = 0; k0 < K; k0 += 32) {
    v16bf af0 = load_frag_a(A + (size_t)(mbase + ln) * K + k0 + hf * 8);
    v16bf af1 = load_frag_a(A + (size_t)(mbase + 16 + ln) * K + k0 + hf * 8);
#pragma unroll
    for (int j = 0; j < 4; ++j) {
      const bf16* bp = Bs + (size_t)(j * 16 + ln) * K + k0 + hf * 8;
      v16bf bfrag = load_frag_a(bp);                  // ds_load_b128 x2
      acc[0][j] = __builtin_amdgcn_wmma_f32_16x16x32_bf16(
          false, af0, false, bfrag, (short)0, acc[0][j], false, false);
      acc[1][j] = __builtin_amdgcn_wmma_f32_16x16x32_bf16(
          false, af1, false, bfrag, (short)0, acc[1][j], false, false);
    }
  }

#pragma unroll
  for (int g = 0; g < 2; ++g) {
#pragma unroll
    for (int j = 0; j < 4; ++j) {
#pragma unroll
      for (int v = 0; v < 8; ++v) {
        int m = mbase + g * 16 + v + 8 * hf;
        int n = nbase + j * 16 + ln;
        float val = acc[g][j][v];
        if (Cf) Cf[(size_t)m * N + n] = val + (bias ? bias[n] : 0.0f);
        else    Cbf[(size_t)m * N + n] = f2bf(val);
      }
    }
  }
}

// ---------------------------------------------------------------------------
// Causal flash attention, one wave per 16-row Q tile, 32 keys per step.
// Q,K in [b*SEQ + s][h*64 + d] bf16; Vt in [b][h][d][s] bf16.
// ---------------------------------------------------------------------------
__global__ __launch_bounds__(128)
void attn_wmma(const bf16* __restrict__ Qb, const bf16* __restrict__ Kb,
               const bf16* __restrict__ Vt, bf16* __restrict__ Ctx) {
  __shared__ bf16 plds[4][16][32];   // per-wave P staging (C-frag -> A-frag)
  const int lane = threadIdx.x & 31;
  const int wave = threadIdx.x >> 5;
  const int hf   = lane >> 4;
  const int ln   = lane & 15;

  const int wid = blockIdx.x * 4 + wave;     // 0 .. 4095
  const int qt  = wid & 127;                 // SEQ/16 = 128 q tiles
  const int h   = (wid >> 7) & 15;
  const int b   = wid >> 11;
  const int qbase = qt * 16;

  // Q fragments for Dh=64 (two K=32 steps), loaded once
  v16bf qf[2];
  {
    const bf16* qrow =
        Qb + ((size_t)(b * SEQ) + qbase + ln) * D_MODEL + h * HDIM;
    qf[0] = load_frag_a(qrow + 0  + hf * 8);
    qf[1] = load_frag_a(qrow + 32 + hf * 8);
  }

  v8f O[4] = {};
  float mrun[8], lrun[8];
#pragma unroll
  for (int v = 0; v < 8; ++v) { mrun[v] = -1e30f; lrun[v] = 0.0f; }

  const float sc = 0.125f;                   // 1/sqrt(64)
  const int t_end = qbase + 16;              // causal: skip fully-masked tiles
  for (int t = 0; t < t_end; t += 32) {
    // scores: two 16x16 tiles over kv = [t, t+16) and [t+16, t+32)
    v8f s[2] = {};
#pragma unroll
    for (int j = 0; j < 2; ++j) {
      const bf16* krow =
          Kb + ((size_t)(b * SEQ) + t + j * 16 + ln) * D_MODEL + h * HDIM;
#pragma unroll
      for (int kk = 0; kk < 2; ++kk) {
        v16bf kfrag = load_frag_a(krow + kk * 32 + hf * 8);
        s[j] = __builtin_amdgcn_wmma_f32_16x16x32_bf16(
            false, qf[kk], false, kfrag, (short)0, s[j], false, false);
      }
    }

    // scale + causal mask + online softmax (row reductions across 16 lanes)
    float pv[2][8];
#pragma unroll
    for (int v = 0; v < 8; ++v) {
      int m = qbase + v + 8 * hf;
#pragma unroll
      for (int j = 0; j < 2; ++j) {
        int kv = t + j * 16 + ln;
        float x = s[j][v] * sc;
        pv[j][v] = (kv <= m) ? x : -1e30f;
      }
    }
#pragma unroll
    for (int v = 0; v < 8; ++v) {
      float mx = fmaxf(pv[0][v], pv[1][v]);
#pragma unroll
      for (int off = 1; off < 16; off <<= 1)
        mx = fmaxf(mx, __shfl_xor(mx, off, 32));
      float mnew  = fmaxf(mrun[v], mx);
      float scale = __expf(mrun[v] - mnew);
      mrun[v] = mnew;
      float p0 = __expf(pv[0][v] - mnew);
      float p1 = __expf(pv[1][v] - mnew);
      pv[0][v] = p0; pv[1][v] = p1;
      float sm = p0 + p1;
#pragma unroll
      for (int off = 1; off < 16; off <<= 1)
        sm += __shfl_xor(sm, off, 32);
      lrun[v] = lrun[v] * scale + sm;
#pragma unroll
      for (int j = 0; j < 4; ++j) O[j][v] *= scale;
    }

    // Stage P (C-fragment layout) through LDS, reload in A-fragment layout
#pragma unroll
    for (int v = 0; v < 8; ++v) {
      plds[wave][v + 8 * hf][ln]      = f2bf(pv[0][v]);
      plds[wave][v + 8 * hf][16 + ln] = f2bf(pv[1][v]);
    }
    asm volatile("s_wait_dscnt 0" ::: "memory");
    v16bf pf = load_frag_a(&plds[wave][ln][hf * 8]);
    asm volatile("" ::: "memory");

    // ctx(16x64) += P(16x32) x V(32x64)
#pragma unroll
    for (int j = 0; j < 4; ++j) {
      const bf16* vrow =
          Vt + ((size_t)(b * NHEAD + h) * HDIM + j * 16 + ln) * SEQ + t + hf * 8;
      v16bf vfrag = load_frag_a(vrow);
      O[j] = __builtin_amdgcn_wmma_f32_16x16x32_bf16(
          false, pf, false, vfrag, (short)0, O[j], false, false);
    }
  }

  // normalize and store ctx (bf16, gemm-input layout)
#pragma unroll
  for (int j = 0; j < 4; ++j) {
#pragma unroll
    for (int v = 0; v < 8; ++v) {
      int m = qbase + v + 8 * hf;
      float val = O[j][v] / lrun[v];
      Ctx[((size_t)(b * SEQ) + m) * D_MODEL + h * HDIM + j * 16 + ln] =
          f2bf(val);
    }
  }
}

// ---------------------------------------------------------------------------
extern "C" void kernel_launch(void* const* d_in, const int* in_sizes, int n_in,
                              void* d_out, int out_size, void* d_ws,
                              size_t ws_size, hipStream_t stream) {
  (void)in_sizes; (void)n_in; (void)out_size; (void)ws_size;
  const float* x  = (const float*)d_in[0];
  const float* Wq = (const float*)d_in[1];
  const float* Wk = (const float*)d_in[2];
  const float* Wv = (const float*)d_in[3];
  const float* Wo = (const float*)d_in[4];
  const float* bo = (const float*)d_in[5];
  float* out = (float*)d_out;

  char* ws = (char*)d_ws;
  const size_t MiB = (size_t)1 << 20;
  bf16* xb  = (bf16*)(ws + 0);          // 8  MiB: x bf16 [4096][1024]
  bf16* wt  = (bf16*)(ws + 8  * MiB);   // 6  MiB: Wq,Wk,Wv transposed bf16
  bf16* wot = (bf16*)(ws + 14 * MiB);   // 2  MiB: Wo transposed bf16
  bf16* qb  = (bf16*)(ws + 16 * MiB);   // 8  MiB
  bf16* kb  = (bf16*)(ws + 24 * MiB);   // 8  MiB
  bf16* vb  = (bf16*)(ws + 32 * MiB);   // 8  MiB
  bf16* vt  = (bf16*)(ws + 40 * MiB);   // 8  MiB: V transposed [b][h][d][s]
  bf16* ctx = (bf16*)(ws + 48 * MiB);   // 8  MiB

  const int nx = MROWS * D_MODEL;       // 4M elements
  const int nw = D_MODEL * D_MODEL;     // 1M elements

  cvt_bf16_kernel<<<(nx + 255) / 256, 256, 0, stream>>>(x, xb, nx);
  packw_kernel<<<(nw + 255) / 256, 256, 0, stream>>>(Wq, wt + 0 * (size_t)nw,
                                                     D_MODEL, D_MODEL);
  packw_kernel<<<(nw + 255) / 256, 256, 0, stream>>>(Wk, wt + 1 * (size_t)nw,
                                                     D_MODEL, D_MODEL);
  packw_kernel<<<(nw + 255) / 256, 256, 0, stream>>>(Wv, wt + 2 * (size_t)nw,
                                                     D_MODEL, D_MODEL);
  packw_kernel<<<(nw + 255) / 256, 256, 0, stream>>>(Wo, wot, D_MODEL, D_MODEL);

  const size_t panel_lds = (size_t)64 * D_MODEL * sizeof(bf16);  // 128 KiB
  dim3 gg(D_MODEL / 64, MROWS / 128);
  gemm_bf16_wmma<<<gg, 128, panel_lds, stream>>>(xb, wt + 0 * (size_t)nw, qb,
                                                 nullptr, nullptr, MROWS,
                                                 D_MODEL, D_MODEL);
  gemm_bf16_wmma<<<gg, 128, panel_lds, stream>>>(xb, wt + 1 * (size_t)nw, kb,
                                                 nullptr, nullptr, MROWS,
                                                 D_MODEL, D_MODEL);
  gemm_bf16_wmma<<<gg, 128, panel_lds, stream>>>(xb, wt + 2 * (size_t)nw, vb,
                                                 nullptr, nullptr, MROWS,
                                                 D_MODEL, D_MODEL);

  transpose_v_kernel<<<(nx + 255) / 256, 256, 0, stream>>>(vb, vt);

  // one wave per 16-row q tile: 2*16*128 = 4096 tiles, 4 waves/block
  attn_wmma<<<(BATCH * NHEAD * (SEQ / 16)) / 4, 128, 0, stream>>>(qb, kb, vt,
                                                                  ctx);

  gemm_bf16_wmma<<<gg, 128, panel_lds, stream>>>(ctx, wot, nullptr, out, bo,
                                                 MROWS, D_MODEL, D_MODEL);
}
